// GGAT_FullMetric_77360950936034
// MI455X (gfx1250) — compile-verified, matrix operand order checked by reading, not run on hardware
//
#include <hip/hip_runtime.h>

#define NN      50000
#define NE      800000
#define IN_F    256
#define HEADS   4
#define D_K     32
#define OUT_T   128
#define N_CAT   384                 // Q|K|V concatenated output cols
#define M_TILES (NN / 16)           // 3125
#define N_TILES (N_CAT / 16)        // 24

typedef __attribute__((ext_vector_type(16))) __bf16   v16bf;
typedef __attribute__((ext_vector_type(8)))  float    v8f;
typedef __attribute__((ext_vector_type(4)))  unsigned uint4v;

union Frag { uint4v q[2]; v16bf v; };

__device__ __forceinline__ unsigned short f2bf(float f) {
    unsigned u = __float_as_uint(f);
    unsigned r = (u + 0x7FFFu + ((u >> 16) & 1u)) >> 16;   // round-to-nearest-even
    return (unsigned short)r;
}
__device__ __forceinline__ float bf2f(unsigned short h) {
    return __uint_as_float(((unsigned)h) << 16);
}

__device__ __forceinline__ unsigned enc_ord(float f) {
    unsigned b = __float_as_uint(f);
    return (b & 0x80000000u) ? ~b : (b | 0x80000000u);
}
__device__ __forceinline__ float dec_ord(unsigned u) {
    return (u & 0x80000000u) ? __uint_as_float(u & 0x7FFFFFFFu) : __uint_as_float(~u);
}

// ---------------------------------------------------------------- utilities
__global__ void k_zero(float* __restrict__ p, int n) {
    int g = blockIdx.x * blockDim.x + threadIdx.x;
    if (g < n) p[g] = 0.0f;
}

// split fp32 -> bf16 hi + bf16 lo (x ~= hi + lo, ~16+ mantissa bits total)
__global__ void k_split(const float* __restrict__ src,
                        unsigned short* __restrict__ hi,
                        unsigned short* __restrict__ lo, int n) {
    int g = blockIdx.x * blockDim.x + threadIdx.x;
    if (g >= n) return;
    float f = src[g];
    unsigned short h = f2bf(f);
    hi[g] = h;
    lo[g] = f2bf(f - bf2f(h));
}

// Wcat^T[n][k] = W_{q|k|v}[k][n%128], split into bf16 hi/lo, [N_CAT x IN_F]
__global__ void k_wcat(const float* __restrict__ Wq, const float* __restrict__ Wk,
                       const float* __restrict__ Wv,
                       unsigned short* __restrict__ whi,
                       unsigned short* __restrict__ wlo) {
    int g = blockIdx.x * blockDim.x + threadIdx.x;
    if (g >= N_CAT * IN_F) return;
    int n = g / IN_F, k = g - n * IN_F;
    int which = n >> 7, nn = n & 127;
    const float* W = (which == 0) ? Wq : (which == 1) ? Wk : Wv;
    float f = W[k * OUT_T + nn];
    unsigned short h = f2bf(f);
    whi[g] = h;
    wlo[g] = f2bf(f - bf2f(h));
}

// Ssym[h][i][j] = 0.5*(S[h][i][j] + S[h][j][i])
__global__ void k_ssym(const float* __restrict__ S, float* __restrict__ out) {
    int g = blockIdx.x * blockDim.x + threadIdx.x;
    if (g >= HEADS * D_K * D_K) return;
    int h = g / (D_K * D_K);
    int r = g - h * D_K * D_K;
    int i = r / D_K, j = r - i * D_K;
    out[g] = 0.5f * (S[h * D_K * D_K + i * D_K + j] + S[h * D_K * D_K + j * D_K + i]);
}

// ------------------------------------------------ fused QKV GEMM via WMMA
// One wave computes one 16x16 tile of [NN x 384]; split-bf16 => 3 WMMAs per
// K-step (hi*hi + hi*lo + lo*hi) for ~fp32 fidelity, fp32 accumulate.
__global__ __launch_bounds__(256) void k_qkv(const unsigned short* __restrict__ xhi,
                                             const unsigned short* __restrict__ xlo,
                                             const unsigned short* __restrict__ whi,
                                             const unsigned short* __restrict__ wlo,
                                             float* __restrict__ Qd,
                                             float* __restrict__ Kd,
                                             float* __restrict__ Vd) {
    int wid = blockIdx.x * 8 + (threadIdx.x >> 5);
    if (wid >= M_TILES * N_TILES) return;          // uniform per wave -> EXEC all-1s
    int lane  = threadIdx.x & 31;
    int mTile = wid / N_TILES;
    int nTile = wid - mTile * N_TILES;
    int l16   = lane & 15;
    int lhalf = lane >> 4;

    size_t aoff = (size_t)(mTile * 16 + l16) * IN_F;
    size_t boff = (size_t)(nTile * 16 + l16) * IN_F;
    const unsigned short* ah = xhi + aoff;
    const unsigned short* al = xlo + aoff;
    const unsigned short* bh = whi + boff;
    const unsigned short* bl = wlo + boff;

    v8f c = {};
#pragma unroll
    for (int ks = 0; ks < IN_F; ks += 32) {
        int c0 = ks + lhalf * 8;                   // lanes0-15: K 0-7 / lanes16-31: K 8-15
        Frag fah, fal, fbh, fbl;
        fah.q[0] = *reinterpret_cast<const uint4v*>(ah + c0);
        fah.q[1] = *reinterpret_cast<const uint4v*>(ah + c0 + 16);
        fbh.q[0] = *reinterpret_cast<const uint4v*>(bh + c0);
        fbh.q[1] = *reinterpret_cast<const uint4v*>(bh + c0 + 16);
        fal.q[0] = *reinterpret_cast<const uint4v*>(al + c0);
        fal.q[1] = *reinterpret_cast<const uint4v*>(al + c0 + 16);
        fbl.q[0] = *reinterpret_cast<const uint4v*>(bl + c0);
        fbl.q[1] = *reinterpret_cast<const uint4v*>(bl + c0 + 16);
        c = __builtin_amdgcn_wmma_f32_16x16x32_bf16(false, fah.v, false, fbh.v,
                                                    (short)0, c, false, false);
        c = __builtin_amdgcn_wmma_f32_16x16x32_bf16(false, fah.v, false, fbl.v,
                                                    (short)0, c, false, false);
        c = __builtin_amdgcn_wmma_f32_16x16x32_bf16(false, fal.v, false, fbh.v,
                                                    (short)0, c, false, false);
    }

    int nBase = nTile * 16;
    float* dst; int ncol;
    if (nBase < 128)      { dst = Qd; ncol = nBase;       }
    else if (nBase < 256) { dst = Kd; ncol = nBase - 128; }
    else                  { dst = Vd; ncol = nBase - 256; }
    int col     = ncol + l16;
    int rowBase = mTile * 16 + lhalf * 8;          // C/D layout: VGPR v -> M = v (+8 hi lanes)
#pragma unroll
    for (int v = 0; v < 8; ++v)
        dst[(size_t)(rowBase + v) * OUT_T + col] = c[v];
}

// ----------------------------------------- edge pass 1: scores + segment max
__global__ __launch_bounds__(256) void k_scores(const float* __restrict__ Qd,
                                                const float* __restrict__ Kd,
                                                const float* __restrict__ Ssym,
                                                const int* __restrict__ ei,
                                                float* __restrict__ scores,
                                                unsigned* __restrict__ m_u) {
    __shared__ float sS[HEADS * D_K * D_K];        // 16 KB of 320 KB LDS
    for (int i = threadIdx.x; i < HEADS * D_K * D_K; i += 256) sS[i] = Ssym[i];
    __syncthreads();

    int g = blockIdx.x * 256 + threadIdx.x;
    if (g >= NE * HEADS) return;
    int e = g >> 2, h = g & 3;
    int row = ei[e];
    int col = ei[NE + e];

    const float* Kp = Kd + ((size_t)row * HEADS + h) * D_K;
    const float* Qp = Qd + ((size_t)col * HEADS + h) * D_K;
    float diff[D_K];
#pragma unroll
    for (int i = 0; i < D_K; ++i) diff[i] = Kp[i] - Qp[i];

    const float* Sh = sS + h * D_K * D_K;
    float s = 0.0f;
    for (int i = 0; i < D_K; ++i) {
        float acc = 0.0f;
#pragma unroll
        for (int j = 0; j < D_K; ++j) acc = fmaf(Sh[i * D_K + j], diff[j], acc);
        s = fmaf(diff[i], acc, s);
    }
    s *= 0.17677669529663687f;                     // 1/sqrt(32)
    s = (s > 0.0f) ? s : 0.2f * s;                 // leaky relu, slope 0.2
    scores[g] = s;
    atomicMax(&m_u[row * HEADS + h], enc_ord(s));
}

// ------------------------------------- edge pass 2: exp + segment sum (denom)
__global__ __launch_bounds__(256) void k_exp(const int* __restrict__ ei,
                                             const unsigned* __restrict__ m_u,
                                             float* __restrict__ scores,
                                             float* __restrict__ denom) {
    int g = blockIdx.x * 256 + threadIdx.x;
    if (g >= NE * HEADS) return;
    int e = g >> 2, h = g & 3;
    int row = ei[e];
    float mf = dec_ord(m_u[row * HEADS + h]);
    float ex = expf(scores[g] - mf);
    scores[g] = ex;
    __hip_atomic_fetch_add(&denom[row * HEADS + h], ex,
                           __ATOMIC_RELAXED, __HIP_MEMORY_SCOPE_AGENT);
}

// ---------------------------------- edge pass 3: alpha * V[col] -> out[row]
__global__ __launch_bounds__(256) void k_acc(const int* __restrict__ ei,
                                             const float* __restrict__ scores,
                                             const float* __restrict__ denom,
                                             const float* __restrict__ Vd,
                                             float* __restrict__ out) {
    int g = blockIdx.x * 256 + threadIdx.x;
    if (g >= NE * HEADS) return;
    int e = g >> 2, h = g & 3;
    int row = ei[e], col = ei[NE + e];
    float alpha = scores[g] / (denom[row * HEADS + h] + 1e-16f);
    const float* Vp = Vd + ((size_t)col * HEADS + h) * D_K;
    float* op = out + (size_t)row * OUT_T + h * D_K;
#pragma unroll
    for (int d = 0; d < D_K; ++d)
        __hip_atomic_fetch_add(&op[d], alpha * Vp[d],
                               __ATOMIC_RELAXED, __HIP_MEMORY_SCOPE_AGENT);
}

// ---------------------------------------------------------------------------
extern "C" void kernel_launch(void* const* d_in, const int* in_sizes, int n_in,
                              void* d_out, int out_size, void* d_ws, size_t ws_size,
                              hipStream_t stream) {
    const float* x  = (const float*)d_in[0];
    const int*   ei = (const int*)  d_in[1];
    const float* Wq = (const float*)d_in[2];
    const float* Wk = (const float*)d_in[3];
    const float* Wv = (const float*)d_in[4];
    const float* Sr = (const float*)d_in[5];
    float* out = (float*)d_out;

    char* ws = (char*)d_ws;
    size_t off = 0;
    auto alloc = [&](size_t bytes) -> void* {
        void* p = ws + off;
        off = (off + bytes + 255) & ~(size_t)255;
        return p;
    };
    float*          Qd  = (float*)         alloc((size_t)NN * OUT_T * 4);
    float*          Kd  = (float*)         alloc((size_t)NN * OUT_T * 4);
    float*          Vd  = (float*)         alloc((size_t)NN * OUT_T * 4);
    unsigned short* xhi = (unsigned short*)alloc((size_t)NN * IN_F * 2);
    unsigned short* xlo = (unsigned short*)alloc((size_t)NN * IN_F * 2);
    unsigned short* whi = (unsigned short*)alloc((size_t)N_CAT * IN_F * 2);
    unsigned short* wlo = (unsigned short*)alloc((size_t)N_CAT * IN_F * 2);
    float*          ssy = (float*)         alloc((size_t)HEADS * D_K * D_K * 4);
    unsigned*       m_u = (unsigned*)      alloc((size_t)NN * HEADS * 4);
    float*          den = (float*)         alloc((size_t)NN * HEADS * 4);
    float*          sc  = (float*)         alloc((size_t)NE * HEADS * 4);

    // zero: out, m_u (bits 0 == encoded minimum), denom
    k_zero<<<(NN * OUT_T + 255) / 256, 256, 0, stream>>>(out, NN * OUT_T);
    k_zero<<<(NN * HEADS + 255) / 256, 256, 0, stream>>>((float*)m_u, NN * HEADS);
    k_zero<<<(NN * HEADS + 255) / 256, 256, 0, stream>>>(den, NN * HEADS);

    // preprocess: split-bf16 operands, W concat+transpose, symmetrized metric
    k_split<<<(NN * IN_F + 255) / 256, 256, 0, stream>>>(x, xhi, xlo, NN * IN_F);
    k_wcat <<<(N_CAT * IN_F + 255) / 256, 256, 0, stream>>>(Wq, Wk, Wv, whi, wlo);
    k_ssym <<<(HEADS * D_K * D_K + 255) / 256, 256, 0, stream>>>(Sr, ssy);

    // fused QKV projection (WMMA, split-bf16): 75000 wave-tiles, 8 waves/block
    k_qkv<<<(M_TILES * N_TILES + 7) / 8, 256, 0, stream>>>(xhi, xlo, whi, wlo, Qd, Kd, Vd);

    // edge passes
    int egrid = (NE * HEADS + 255) / 256;
    k_scores<<<egrid, 256, 0, stream>>>(Qd, Kd, ssy, ei, sc, m_u);
    k_exp   <<<egrid, 256, 0, stream>>>(ei, m_u, sc, den);
    k_acc   <<<egrid, 256, 0, stream>>>(ei, sc, den, Vd, out);
}
